// SDDRNYURefiner_85023172591726
// MI455X (gfx1250) — compile-verified, matrix operand order checked by reading, not run on hardware
//
#include <hip/hip_runtime.h>
#include <hip/hip_fp16.h>
#include <cmath>

// ---------------------------------------------------------------------------
// Types for WMMA
// ---------------------------------------------------------------------------
typedef __attribute__((ext_vector_type(16))) _Float16 v16h;
typedef __attribute__((ext_vector_type(8)))  _Float16 v8h;
typedef __attribute__((ext_vector_type(8)))  float    v8f;
typedef __attribute__((ext_vector_type(4)))  int      v4i;

union FragH { v16h v; v8h q[2]; _Float16 h[16]; };
union FragF { v8f  v; float f[8]; };

#if defined(__gfx1250__) && __has_builtin(__builtin_amdgcn_global_load_async_to_lds_b128) && __has_builtin(__builtin_amdgcn_s_wait_asynccnt)
#define ASYNC_LDS 1
#endif

// ---------------------------------------------------------------------------
// Device reduction helpers (wave32)
// ---------------------------------------------------------------------------
__device__ __forceinline__ float warp_sum(float v) {
#pragma unroll
  for (int o = 16; o > 0; o >>= 1) v += __shfl_down(v, o, 32);
  return v;
}
__device__ __forceinline__ float warp_max(float v) {
#pragma unroll
  for (int o = 16; o > 0; o >>= 1) v = fmaxf(v, __shfl_down(v, o, 32));
  return v;
}
__device__ float block_sum(float v, float* sbuf) {
  const int lane = threadIdx.x & 31, w = threadIdx.x >> 5;
  v = warp_sum(v);
  __syncthreads();
  if (lane == 0) sbuf[w] = v;
  __syncthreads();
  const int nw = (blockDim.x + 31) >> 5;
  float r = ((int)threadIdx.x < nw) ? sbuf[threadIdx.x] : 0.f;
  if (w == 0) { r = warp_sum(r); if (lane == 0) sbuf[0] = r; }
  __syncthreads();
  float out = sbuf[0];
  __syncthreads();
  return out;
}
__device__ float block_max(float v, float* sbuf) {
  const int lane = threadIdx.x & 31, w = threadIdx.x >> 5;
  v = warp_max(v);
  __syncthreads();
  if (lane == 0) sbuf[w] = v;
  __syncthreads();
  const int nw = (blockDim.x + 31) >> 5;
  float r = ((int)threadIdx.x < nw) ? sbuf[threadIdx.x] : -3.0e38f;
  if (w == 0) { r = warp_max(r); if (lane == 0) sbuf[0] = r; }
  __syncthreads();
  float out = sbuf[0];
  __syncthreads();
  return out;
}

// ---------------------------------------------------------------------------
// WMMA GEMM (TN): C[M,N] = A[M,K] * Bt[N,K]^T, f16 operands, f32 out.
// K (lda/ldb) must be a multiple of 32 halves and zero-padded; bases 16B
// aligned. Epilogue: v = acc*scale[m] + bias[m|col] + res[m*ldres+col]; relu.
// Block: 256 threads = 8 waves; wave -> 16x32 C tile; block tile 64x64, K=32.
// ---------------------------------------------------------------------------
__device__ __forceinline__ void gemm_store(
    float acc, int m, int col, int M, int N, float* Cp, int ldc,
    const float* scale, const float* bias, int bias_mode,
    const float* res, int ldres, int relu) {
  if (m >= M || col >= N) return;
  float v = acc;
  if (scale) v *= scale[m];
  if (bias)  v += (bias_mode == 2) ? bias[col] : bias[m];
  if (res)   v += res[(size_t)m * ldres + col];
  if (relu)  v = fmaxf(v, 0.f);
  Cp[(size_t)m * ldc + col] = v;
}

__global__ __launch_bounds__(256)
void k_wmma_gemm(const _Float16* __restrict__ A, int lda,
                 const _Float16* __restrict__ B, int ldb,
                 float* __restrict__ Cp, int ldc,
                 int M, int N, int K,
                 const float* __restrict__ scale,
                 const float* __restrict__ bias, int bias_mode,
                 const float* __restrict__ res, int ldres, int relu) {
  __shared__ _Float16 As[64][40];   // row stride 80B -> 16B aligned chunks
  __shared__ _Float16 Bs[64][40];
  const int tid  = threadIdx.x;
  const int bm   = blockIdx.y * 64;
  const int bn   = blockIdx.x * 64;
  const int wave = tid >> 5;
  const int lane = tid & 31;
  const int wm   = wave >> 1;   // 0..3  (M sub-tile)
  const int wn   = wave & 1;    // 0..1  (N sub-tile, 32 wide)
  const int lrow = lane & 15;
  const int lhalf = lane >> 4;

  const int r  = tid >> 2;          // 0..63 (staging row)
  const int c0 = (tid & 3) << 3;    // 0,8,16,24 (staging K chunk)
  const int gm = bm + r;
  const int gn = bn + r;
  const bool fullT = (bm + 64 <= M) && (bn + 64 <= N);

  FragF acc0 = {}; FragF acc1 = {};

  for (int k0 = 0; k0 < K; k0 += 32) {
    if (fullT) {
#if ASYNC_LDS
      __builtin_amdgcn_global_load_async_to_lds_b128(
          (v4i*)(A + (size_t)gm * lda + k0 + c0),
          (v4i*)&As[r][c0], 0, 0);
      __builtin_amdgcn_global_load_async_to_lds_b128(
          (v4i*)(B + (size_t)gn * ldb + k0 + c0),
          (v4i*)&Bs[r][c0], 0, 0);
#else
      v8h va = *(const v8h*)(A + (size_t)gm * lda + k0 + c0);
      v8h vb = *(const v8h*)(B + (size_t)gn * ldb + k0 + c0);
      *(v8h*)&As[r][c0] = va;
      *(v8h*)&Bs[r][c0] = vb;
#endif
    } else {
      v8h va = {};
      v8h vb = {};
      if (gm < M) va = *(const v8h*)(A + (size_t)gm * lda + k0 + c0);
      if (gn < N) vb = *(const v8h*)(B + (size_t)gn * ldb + k0 + c0);
      *(v8h*)&As[r][c0] = va;
      *(v8h*)&Bs[r][c0] = vb;
    }
    if (k0 + 32 < K) { // prefetch next K-tile (global_prefetch_b8)
      int pm = gm < M ? gm : M - 1;
      int pn = gn < N ? gn : N - 1;
      __builtin_prefetch(&A[(size_t)pm * lda + k0 + 32], 0, 1);
      __builtin_prefetch(&B[(size_t)pn * ldb + k0 + 32], 0, 1);
    }
#if ASYNC_LDS
    if (fullT) __builtin_amdgcn_s_wait_asynccnt(0);
#endif
    __syncthreads();
    // ISA 16-bit fragment striping: lane-half h<8 -> K in [lhalf*8, +8),
    // h>=8 -> K in [16+lhalf*8, +8). Both are contiguous 16B LDS chunks.
    FragH fa, fb0, fb1;
    fa.q[0]  = *(const v8h*)&As[wm * 16 + lrow][lhalf * 8];
    fa.q[1]  = *(const v8h*)&As[wm * 16 + lrow][16 + lhalf * 8];
    fb0.q[0] = *(const v8h*)&Bs[wn * 32 + lrow][lhalf * 8];
    fb0.q[1] = *(const v8h*)&Bs[wn * 32 + lrow][16 + lhalf * 8];
    fb1.q[0] = *(const v8h*)&Bs[wn * 32 + 16 + lrow][lhalf * 8];
    fb1.q[1] = *(const v8h*)&Bs[wn * 32 + 16 + lrow][16 + lhalf * 8];
    acc0.v = __builtin_amdgcn_wmma_f32_16x16x32_f16(false, fa.v, false, fb0.v,
                                                    (short)0, acc0.v, false, false);
    acc1.v = __builtin_amdgcn_wmma_f32_16x16x32_f16(false, fa.v, false, fb1.v,
                                                    (short)0, acc1.v, false, false);
    __syncthreads();
  }
  const int mrow = bm + wm * 16 + lhalf * 8;
  const int col0 = bn + wn * 32 + lrow;
#pragma unroll
  for (int rr = 0; rr < 8; ++rr) {
    gemm_store(acc0.f[rr], mrow + rr, col0,      M, N, Cp, ldc, scale, bias, bias_mode, res, ldres, relu);
    gemm_store(acc1.f[rr], mrow + rr, col0 + 16, M, N, Cp, ldc, scale, bias, bias_mode, res, ldres, relu);
  }
}

// ---------------------------------------------------------------------------
// Packing kernels (all produce K-major f16, zero-padded to Kp)
// ---------------------------------------------------------------------------
// dst[r*Kp + k] = k<K ? src[r*K + k] : 0
__global__ void k_f16_pad(_Float16* dst, const float* src, int rows, int K, int Kp) {
  long long tot = (long long)rows * Kp;
  for (long long i = blockIdx.x * (long long)blockDim.x + threadIdx.x; i < tot;
       i += (long long)gridDim.x * blockDim.x) {
    int k = (int)(i % Kp), rr = (int)(i / Kp);
    dst[i] = (k < K) ? (_Float16)src[(size_t)rr * K + k] : (_Float16)0.0f;
  }
}
// dst[t*dhp + d] = d<dh ? src[t*ld + col0 + d] : 0
__global__ void k_pack_rows(_Float16* dst, const float* src, int S, int ld,
                            int col0, int dh, int dhp) {
  long long tot = (long long)S * dhp;
  for (long long i = blockIdx.x * (long long)blockDim.x + threadIdx.x; i < tot;
       i += (long long)gridDim.x * blockDim.x) {
    int d = (int)(i % dhp), t = (int)(i / dhp);
    dst[i] = (d < dh) ? (_Float16)src[(size_t)t * ld + col0 + d] : (_Float16)0.0f;
  }
}
// dst[d*Sp + t] = t<S ? src[t*ld + col0 + d] : 0
__global__ void k_pack_t_pad(_Float16* dst, const float* src, int S, int Sp,
                             int ld, int col0, int dh) {
  long long tot = (long long)dh * Sp;
  for (long long i = blockIdx.x * (long long)blockDim.x + threadIdx.x; i < tot;
       i += (long long)gridDim.x * blockDim.x) {
    int t = (int)(i % Sp), d = (int)(i / Sp);
    dst[i] = (t < S) ? (_Float16)src[(size_t)t * ld + col0 + d] : (_Float16)0.0f;
  }
}
// im2col, pixel-major: dst[n*Kp + k], zero pad k >= K
__global__ void k_im2col_f16(_Float16* dst, const float* src, int Cin, int Hin,
                             int Win, int kh, int kw, int stride, int pad,
                             int Ho, int Wo, int K, int Kp) {
  long long tot = (long long)Ho * Wo * Kp;
  for (long long idx = blockIdx.x * (long long)blockDim.x + threadIdx.x;
       idx < tot; idx += (long long)gridDim.x * blockDim.x) {
    int k = (int)(idx % Kp), n = (int)(idx / Kp);
    float v = 0.f;
    if (k < K) {
      int c = k / (kh * kw), rr = k % (kh * kw);
      int dy = rr / kw, dx = rr % kw;
      int oy = n / Wo, ox = n % Wo;
      int iy = oy * stride - pad + dy, ix = ox * stride - pad + dx;
      if (iy >= 0 && iy < Hin && ix >= 0 && ix < Win)
        v = src[((size_t)c * Hin + iy) * Win + ix];
    }
    dst[idx] = (_Float16)v;
  }
}

// ---------------------------------------------------------------------------
// Elementwise / data-movement kernels
// ---------------------------------------------------------------------------
__global__ void k_maxpool3s2(float* out, const float* in, int C, int H, int W,
                             int Ho, int Wo) {
  long long tot = (long long)C * Ho * Wo;
  for (long long idx = blockIdx.x * (long long)blockDim.x + threadIdx.x;
       idx < tot; idx += (long long)gridDim.x * blockDim.x) {
    int c = (int)(idx / ((long long)Ho * Wo));
    int rr = (int)(idx % ((long long)Ho * Wo));
    int oy = rr / Wo, ox = rr % Wo;
    float m = -3.0e38f;
    for (int dy = 0; dy < 3; ++dy) {
      int iy = oy * 2 - 1 + dy; if (iy < 0 || iy >= H) continue;
      for (int dx = 0; dx < 3; ++dx) {
        int ix = ox * 2 - 1 + dx; if (ix < 0 || ix >= W) continue;
        m = fmaxf(m, in[((size_t)c * H + iy) * W + ix]);
      }
    }
    out[idx] = m;
  }
}
// bilinear, half-pixel centers
__global__ void k_resize(float* out, const float* in, int C, int Hs, int Wsrc,
                         int Hd, int Wd) {
  long long tot = (long long)C * Hd * Wd;
  for (long long idx = blockIdx.x * (long long)blockDim.x + threadIdx.x;
       idx < tot; idx += (long long)gridDim.x * blockDim.x) {
    int c = (int)(idx / ((long long)Hd * Wd));
    int rr = (int)(idx % ((long long)Hd * Wd));
    int y = rr / Wd, x = rr % Wd;
    float fy = ((y + 0.5f) * Hs) / Hd - 0.5f;
    float fx = ((x + 0.5f) * Wsrc) / Wd - 0.5f;
    fy = fminf(fmaxf(fy, 0.f), (float)(Hs - 1));
    fx = fminf(fmaxf(fx, 0.f), (float)(Wsrc - 1));
    int y0 = (int)fy, x0 = (int)fx;
    int y1 = (y0 + 1 < Hs) ? y0 + 1 : Hs - 1;
    int x1 = (x0 + 1 < Wsrc) ? x0 + 1 : Wsrc - 1;
    float wy = fy - y0, wx = fx - x0;
    const float* pc = in + (size_t)c * Hs * Wsrc;
    float v0 = (1.f - wx) * pc[(size_t)y0 * Wsrc + x0] + wx * pc[(size_t)y0 * Wsrc + x1];
    float v1 = (1.f - wx) * pc[(size_t)y1 * Wsrc + x0] + wx * pc[(size_t)y1 * Wsrc + x1];
    out[idx] = (1.f - wy) * v0 + wy * v1;
  }
}
__global__ void k_pack6(float* dst, const float* img, const float* x4,
                        const float* sp, int HW) {
  long long tot = 6LL * HW;
  for (long long idx = blockIdx.x * (long long)blockDim.x + threadIdx.x;
       idx < tot; idx += (long long)gridDim.x * blockDim.x) {
    int ch = (int)(idx / HW), p = (int)(idx % HW);
    float v;
    if (ch < 3) v = img[idx];
    else if (ch == 3) v = x4[p];
    else if (ch == 4) v = sp[p];
    else v = (sp[p] > 0.f) ? 1.f : 0.f;
    dst[idx] = v;
  }
}
__global__ void k_chw_to_tokens(float* x, const float* xl, const float* xh,
                                int C, int HW) {
  long long tot = 2LL * HW * C;
  for (long long idx = blockIdx.x * (long long)blockDim.x + threadIdx.x;
       idx < tot; idx += (long long)gridDim.x * blockDim.x) {
    int t = (int)(idx / C), c = (int)(idx % C);
    x[idx] = (t < HW) ? xl[(size_t)c * HW + t] : xh[(size_t)c * HW + (t - HW)];
  }
}
__global__ void k_tokens_to_chw(float* xl2, float* xh2, const float* x, int C,
                                int HW) {
  long long tot = 2LL * HW * C;
  for (long long idx = blockIdx.x * (long long)blockDim.x + threadIdx.x;
       idx < tot; idx += (long long)gridDim.x * blockDim.x) {
    int t = (int)(idx / C), c = (int)(idx % C);
    if (t < HW) xl2[(size_t)c * HW + t] = x[idx];
    else        xh2[(size_t)c * HW + (t - HW)] = x[idx];
  }
}
__global__ __launch_bounds__(256)
void k_layernorm(float* out, const float* in, int S, int C, const float* g,
                 const float* b) {
  __shared__ float sred[32];
  int row = blockIdx.x; if (row >= S) return;
  const float* x = in + (size_t)row * C;
  float sl = 0.f;
  for (int c = threadIdx.x; c < C; c += blockDim.x) sl += x[c];
  float mu = block_sum(sl, sred) / C;
  float vl = 0.f;
  for (int c = threadIdx.x; c < C; c += blockDim.x) { float d = x[c] - mu; vl += d * d; }
  float var = block_sum(vl, sred) / C;
  float inv = rsqrtf(var + 1e-5f);
  for (int c = threadIdx.x; c < C; c += blockDim.x)
    out[(size_t)row * C + c] = (x[c] - mu) * inv * g[c] + b[c];
}
__global__ __launch_bounds__(256)
void k_softmax(float* att, int S, float scale) {
  __shared__ float sred[32];
  int row = blockIdx.x; if (row >= S) return;
  float* a = att + (size_t)row * S;
  float mx = -3.0e38f;
  for (int i = threadIdx.x; i < S; i += blockDim.x) mx = fmaxf(mx, a[i] * scale);
  mx = block_max(mx, sred);
  float s = 0.f;
  for (int i = threadIdx.x; i < S; i += blockDim.x) {
    float e = __expf(a[i] * scale - mx); a[i] = e; s += e;
  }
  s = block_sum(s, sred);
  float inv = 1.f / s;
  for (int i = threadIdx.x; i < S; i += blockDim.x) a[i] *= inv;
}
__global__ void k_gate(float* out, const float* l2, const float* h2,
                       const float* wv, const float* bv, int C, int HW) {
  int p = blockIdx.x * blockDim.x + threadIdx.x;
  if (p >= HW) return;
  float acc = bv[0];
  for (int c = 0; c < C; ++c) acc += wv[c] * (l2[(size_t)c * HW + p] + h2[(size_t)c * HW + p]);
  float g = 1.f / (1.f + __expf(-acc));
  for (int c = 0; c < C; ++c)
    out[(size_t)c * HW + p] = g * l2[(size_t)c * HW + p] + (1.f - g) * h2[(size_t)c * HW + p];
}

// ---------------------------------------------------------------------------
// Poisson depth completion: one persistent workgroup, whole 192x192 grid
// ping-pongs inside 2x147KB of CDNA5's 320KB LDS.
// ---------------------------------------------------------------------------
#define PN  192
#define PNP (PN * PN)

__global__ __launch_bounds__(1024)
void k_poisson(const float* __restrict__ sp, const float* __restrict__ en,
               float* __restrict__ eb, float* __restrict__ kn,
               float* __restrict__ vk, float* __restrict__ out) {
  extern __shared__ float pls[];
  float* va = pls; float* vb = pls + PNP;
  __shared__ float sred[32];
  const int tid = threadIdx.x, T = blockDim.x;

  float nl = 0.f, sel = 0.f, sdl = 0.f;
  for (int i = tid; i < PNP; i += T) {
    float s = sp[i]; float mm = (s > 0.f) ? 1.f : 0.f;
    nl += mm; sel += mm * en[i]; sdl += mm * (s * 0.1f);
  }
  float n = block_sum(nl, sred);
  float nm = fmaxf(n, 1.f);
  float me = block_sum(sel, sred) / nm;
  float md = block_sum(sdl, sred) / nm;
  float cvl = 0.f, vel = 0.f, vdl = 0.f;
  for (int i = tid; i < PNP; i += T) {
    float s = sp[i]; float mm = (s > 0.f) ? 1.f : 0.f;
    float de = en[i] - me, dd = s * 0.1f - md;
    cvl += mm * de * dd; vel += mm * de * de; vdl += mm * dd * dd;
  }
  float cov = block_sum(cvl, sred) / nm;
  float se = sqrtf(block_sum(vel, sred) / nm);
  float sd = sqrtf(block_sum(vdl, sred) / nm);
  float corr = cov / (se * sd + 1e-12f);
  int flip = (n >= 10.f) && (corr < 0.f);
  for (int i = tid; i < PNP; i += T) {
    float e = en[i]; if (flip) e = 1.f - e;
    eb[i] = e;
    kn[i] = (sp[i] > 0.f) ? 1.f : 0.f;   // kn doubles as IRLS weight
  }
  __syncthreads();

  float a = 1.f, b0 = 0.f;
  for (int it = 0; it < 3; ++it) {
    float A00 = 0.f, A01 = 0.f, A11 = 0.f, r0 = 0.f, r1 = 0.f;
    for (int i = tid; i < PNP; i += T) {
      float w = kn[i], w2 = w * w, x = eb[i], y = sp[i] * 0.1f;
      A00 += w2 * x * x; A01 += w2 * x; A11 += w2; r0 += w2 * x * y; r1 += w2 * y;
    }
    A00 = block_sum(A00, sred); A01 = block_sum(A01, sred); A11 = block_sum(A11, sred);
    r0 = block_sum(r0, sred);   r1 = block_sum(r1, sred);
    float det = A00 * A11 - A01 * A01 + 1e-12f;
    a  = (r0 * A11 - r1 * A01) / det;
    b0 = (A00 * r1 - A01 * r0) / det;
    float mx = 0.f;
    for (int i = tid; i < PNP; i += T)
      if (sp[i] > 0.f) mx = fmaxf(mx, fabsf(a * eb[i] + b0 - sp[i] * 0.1f));
    mx = block_max(mx, sred);
    float lo = 0.f, hi = mx;
    float target = 0.5f * (n + 1.f);
    for (int bi = 0; bi < 18; ++bi) {
      float mid = 0.5f * (lo + hi);
      float cl = 0.f;
      for (int i = tid; i < PNP; i += T)
        if (sp[i] > 0.f && fabsf(a * eb[i] + b0 - sp[i] * 0.1f) <= mid) cl += 1.f;
      float cnt = block_sum(cl, sred);
      if (cnt >= target) hi = mid; else lo = mid;
    }
    float med = 0.5f * (lo + hi);
    float cc = 1.345f * med + 1e-6f;
    for (int i = tid; i < PNP; i += T) {
      float mm = (sp[i] > 0.f) ? 1.f : 0.f;
      float res = fabsf(a * eb[i] + b0 - sp[i] * 0.1f);
      kn[i] = mm / fmaxf(1.f, res / cc);
    }
    __syncthreads();
  }
  if (n >= 10.f)
    for (int i = tid; i < PNP; i += T)
      eb[i] = fminf(fmaxf(a * eb[i] + b0, 0.f), 1.f);
  __syncthreads();

  for (int i = tid; i < PNP; i += T) {
    int y = i / PN, x = i - y * PN;
    float s9 = 0.f;
    for (int dy = -1; dy <= 1; ++dy) {
      int yy = y + dy; if (yy < 0) yy = -yy; if (yy > PN - 1) yy = 2 * (PN - 1) - yy;
      for (int dx = -1; dx <= 1; ++dx) {
        int xx = x + dx; if (xx < 0) xx = -xx; if (xx > PN - 1) xx = 2 * (PN - 1) - xx;
        s9 += eb[yy * PN + xx];
      }
    }
    float estm = (s9 / 9.f) * 10.f;
    float s = sp[i];
    bool m = s > 0.f;
    vk[i] = m ? s : estm;
    bool border = (y == 0) || (y == PN - 1) || (x == 0) || (x == PN - 1);
    va[i] = estm;
    kn[i] = (m || border) ? 1.f : 0.f;
  }
  __syncthreads();

  float* pa = va; float* pb = vb;
  for (int it = 0; it < 1000; ++it) {
    float dl = 0.f;
    for (int i = tid; i < PNP; i += T) {
      int y = i / PN, x = i - y * PN;
      int yu = (y == 0) ? (PN - 1) : (y - 1), yd = (y == PN - 1) ? 0 : (y + 1);
      int xl = (x == 0) ? (PN - 1) : (x - 1), xr = (x == PN - 1) ? 0 : (x + 1);
      float vn = 0.25f * (pa[yu * PN + x] + pa[yd * PN + x] +
                          pa[y * PN + xl] + pa[y * PN + xr]);
      if (kn[i] > 0.5f) vn = vk[i];
      pb[i] = vn;
      dl += fabsf(vn - pa[i]);
    }
    float dsum = block_sum(dl, sred);
    float* t = pa; pa = pb; pb = t;
    if (dsum < (1e-5f * (float)PNP)) break;
  }
  for (int i = tid; i < PNP; i += T) out[i] = pa[i];
}

// ---------------------------------------------------------------------------
// Host orchestration
// ---------------------------------------------------------------------------
struct Stack { char* base; size_t off; size_t cap; };
static inline void* salloc(Stack& s, size_t bytes) {
  size_t o = (s.off + 255) & ~(size_t)255;
  s.off = o + bytes;
  return s.base + o;
}
static inline int rup32(int x) { return (x + 31) & ~31; }
#define PF(i) ((const float*)din[(i)])
#define GRID1D(n) dim3((unsigned)((((long long)(n)) + 255) / 256))

static void gemm_launch(hipStream_t st, const _Float16* A, int lda,
                        const _Float16* B, int ldb, float* C, int ldc, int M,
                        int N, int K, const float* scale, const float* bias,
                        int bias_mode, const float* res, int ldres, int relu) {
  dim3 g((N + 63) / 64, (M + 63) / 64), b(256);
  k_wmma_gemm<<<g, b, 0, st>>>(A, lda, B, ldb, C, ldc, M, N, K, scale, bias,
                               bias_mode, res, ldres, relu);
}

static float* conv2d(Stack& ws, hipStream_t st, const float* x, int Cin, int H,
                     int W, const float* wgt, int Cout, int kh, int kw,
                     int stride, int pad, const float* gsc, const float* bsc,
                     int bias_mode, int relu, const float* res, int* pHo,
                     int* pWo, float* out) {
  int Ho = (H + 2 * pad - kh) / stride + 1;
  int Wo = (W + 2 * pad - kw) / stride + 1;
  int K = Cin * kh * kw, Kp = rup32(K), N = Ho * Wo;
  if (!out) out = (float*)salloc(ws, (size_t)Cout * N * 4);
  size_t mark = ws.off;
  _Float16* col = (_Float16*)salloc(ws, (size_t)N * Kp * 2);
  _Float16* w16 = (_Float16*)salloc(ws, (size_t)Cout * Kp * 2);
  k_f16_pad<<<GRID1D((long long)Cout * Kp), 256, 0, st>>>(w16, wgt, Cout, K, Kp);
  k_im2col_f16<<<GRID1D((long long)N * Kp), 256, 0, st>>>(col, x, Cin, H, W, kh, kw, stride, pad, Ho, Wo, K, Kp);
  gemm_launch(st, w16, Kp, col, Kp, out, N, Cout, N, Kp, gsc, bsc, bias_mode, res, N, relu);
  ws.off = mark;
  if (pHo) *pHo = Ho;
  if (pWo) *pWo = Wo;
  return out;
}

static float* linear_fw(Stack& ws, hipStream_t st, const float* X, int S,
                        int Cin, int Cout, const float* Wf, const float* bf,
                        int relu, const float* res) {
  int Kp = rup32(Cin);
  float* out = (float*)salloc(ws, (size_t)S * Cout * 4);
  size_t mark = ws.off;
  _Float16* X16 = (_Float16*)salloc(ws, (size_t)S * Kp * 2);
  _Float16* W16 = (_Float16*)salloc(ws, (size_t)Cout * Kp * 2);
  k_f16_pad<<<GRID1D((long long)S * Kp), 256, 0, st>>>(X16, X, S, Cin, Kp);
  k_f16_pad<<<GRID1D((long long)Cout * Kp), 256, 0, st>>>(W16, Wf, Cout, Cin, Kp);
  gemm_launch(st, X16, Kp, W16, Kp, out, Cout, S, Cout, Kp, nullptr, bf,
              bf ? 2 : 0, res, Cout, relu);
  ws.off = mark;
  return out;
}

static float* basic_block(Stack& ws, hipStream_t st, void* const* din,
                          float* x, int Cin, int& H, int& W, int Cout,
                          bool down, int& cur) {
  const float* bn1b = PF(cur + 0); const float* bn1g = PF(cur + 1);
  const float* bn2b = PF(cur + 2); const float* bn2g = PF(cur + 3);
  const float *bndb = nullptr, *bndg = nullptr, *w1, *w2, *wd = nullptr;
  if (down) { bndb = PF(cur + 4); bndg = PF(cur + 5); w1 = PF(cur + 6); w2 = PF(cur + 7); wd = PF(cur + 8); cur += 9; }
  else      { w1 = PF(cur + 4); w2 = PF(cur + 5); cur += 6; }
  int stride = down ? 2 : 1, Ho, Wo;
  float* y1 = conv2d(ws, st, x, Cin, H, W, w1, Cout, 3, 3, stride, 1, bn1g, bn1b, 1, 1, nullptr, &Ho, &Wo, nullptr);
  const float* idn = x;
  if (down)
    idn = conv2d(ws, st, x, Cin, H, W, wd, Cout, 1, 1, stride, 0, bndg, bndb, 1, 0, nullptr, nullptr, nullptr, nullptr);
  float* y2 = conv2d(ws, st, y1, Cout, Ho, Wo, w2, Cout, 3, 3, 1, 1, bn2g, bn2b, 1, 1, idn, nullptr, nullptr, nullptr);
  H = Ho; W = Wo;
  return y2;
}

static void run_encoder(Stack& ws, hipStream_t st, void* const* din, int ENC,
                        const float* z, float* c[4]) {
  int cur = ENC;
  const float* bn1b = PF(cur + 0); const float* bn1g = PF(cur + 1);
  const float* c1w  = PF(cur + 2); cur += 3;
  int H, W;
  float* x = conv2d(ws, st, z, 3, 192, 192, c1w, 64, 7, 7, 2, 3, bn1g, bn1b, 1, 1, nullptr, &H, &W, nullptr);
  float* xp = (float*)salloc(ws, (size_t)64 * 48 * 48 * 4);
  k_maxpool3s2<<<GRID1D(64LL * 48 * 48), 256, 0, st>>>(xp, x, 64, H, W, 48, 48);
  H = W = 48;
  int Cin = 64;
  for (int b = 0; b < 3; ++b) xp = basic_block(ws, st, din, xp, Cin, H, W, 64, false, cur);
  c[0] = xp;
  for (int b = 0; b < 4; ++b) { xp = basic_block(ws, st, din, xp, Cin, H, W, 128, b == 0, cur); Cin = 128; }
  c[1] = xp;
  for (int b = 0; b < 6; ++b) { xp = basic_block(ws, st, din, xp, Cin, H, W, 256, b == 0, cur); Cin = 256; }
  c[2] = xp;
  for (int b = 0; b < 3; ++b) { xp = basic_block(ws, st, din, xp, Cin, H, W, 512, b == 0, cur); Cin = 512; }
  c[3] = xp;
}

static void attn_block(Stack& ws, hipStream_t st, void* const* din, int base,
                       const float* fl, const float* fh, int C, int s,
                       float* outL, float* outH) {
  const float* b1v = PF(base + 0); const float* b2v = PF(base + 1);
  const float* binv = PF(base + 2); const float* bov = PF(base + 3);
  const float* lnb = PF(base + 4); const float* lng = PF(base + 5);
  const float* phw = PF(base + 6); const float* plw = PF(base + 7);
  const float* w1v = PF(base + 8); const float* w2v = PF(base + 9);
  const float* winv = PF(base + 10); const float* wov = PF(base + 11);
  int HW = s * s, S = 2 * HW, dh = C / 4;
  int Sp = rup32(S), dhp = rup32(dh);
  float* xl = conv2d(ws, st, fl, C, s, s, plw, C, 1, 1, 1, 0, nullptr, nullptr, 0, 0, nullptr, nullptr, nullptr, nullptr);
  float* xh = conv2d(ws, st, fh, C, s, s, phw, C, 1, 1, 1, 0, nullptr, nullptr, 0, 0, nullptr, nullptr, nullptr, nullptr);
  float* x = (float*)salloc(ws, (size_t)S * C * 4);
  k_chw_to_tokens<<<GRID1D((long long)S * C), 256, 0, st>>>(x, xl, xh, C, HW);
  float* xln = (float*)salloc(ws, (size_t)S * C * 4);
  k_layernorm<<<dim3(S), dim3(256), 0, st>>>(xln, x, S, C, lng, lnb);
  float* qkv = linear_fw(ws, st, xln, S, C, 3 * C, winv, binv, 0, nullptr);
  _Float16* qbuf = (_Float16*)salloc(ws, (size_t)S * dhp * 2);
  _Float16* kbuf = (_Float16*)salloc(ws, (size_t)S * dhp * 2);
  _Float16* vT   = (_Float16*)salloc(ws, (size_t)dh * Sp * 2);
  float* att = (float*)salloc(ws, (size_t)S * S * 4);
  _Float16* att16 = (_Float16*)salloc(ws, (size_t)S * Sp * 2);
  float* o = (float*)salloc(ws, (size_t)S * C * 4);
  float sc = 1.0f / sqrtf((float)dh);
  for (int h = 0; h < 4; ++h) {
    k_pack_rows<<<GRID1D((long long)S * dhp), 256, 0, st>>>(qbuf, qkv, S, 3 * C, h * dh, dh, dhp);
    k_pack_rows<<<GRID1D((long long)S * dhp), 256, 0, st>>>(kbuf, qkv, S, 3 * C, C + h * dh, dh, dhp);
    gemm_launch(st, qbuf, dhp, kbuf, dhp, att, S, S, S, dhp, nullptr, nullptr, 0, nullptr, 0, 0);
    k_softmax<<<dim3(S), dim3(256), 0, st>>>(att, S, sc);
    k_f16_pad<<<GRID1D((long long)S * Sp), 256, 0, st>>>(att16, att, S, S, Sp);
    k_pack_t_pad<<<GRID1D((long long)dh * Sp), 256, 0, st>>>(vT, qkv, S, Sp, 3 * C, 2 * C + h * dh, dh);
    gemm_launch(st, att16, Sp, vT, Sp, o + h * dh, C, S, dh, Sp, nullptr, nullptr, 0, nullptr, 0, 0);
  }
  float* x2 = linear_fw(ws, st, o, S, C, C, wov, bov, 0, xln);
  float* hb = (float*)salloc(ws, (size_t)S * C * 4);
  k_layernorm<<<dim3(S), dim3(256), 0, st>>>(hb, x2, S, C, lng, lnb);
  float* t = linear_fw(ws, st, hb, S, C, 4 * C, w1v, b1v, 1, nullptr);
  float* x3 = linear_fw(ws, st, t, S, 4 * C, C, w2v, b2v, 0, x2);
  k_tokens_to_chw<<<GRID1D((long long)S * C), 256, 0, st>>>(outL, outH, x3, C, HW);
}

static float* ffm_fw(Stack& ws, hipStream_t st, void* const* din, int base,
                     const float* fc, int Cc, const float* fp, int Cp, int s,
                     int Co) {
  size_t HWl = (size_t)s * s;
  float* cat = (float*)salloc(ws, 2 * Co * HWl * 4);
  conv2d(ws, st, fc, Cc, s, s, PF(base + 2), Co, 3, 3, 1, 1, PF(base + 1), PF(base + 0), 1, 1, nullptr, nullptr, nullptr, cat);
  conv2d(ws, st, fp, Cp, s, s, PF(base + 8), Co, 3, 3, 1, 1, PF(base + 7), PF(base + 6), 1, 1, nullptr, nullptr, nullptr, cat + Co * HWl);
  return conv2d(ws, st, cat, 2 * Co, s, s, PF(base + 5), Co, 3, 3, 1, 1, PF(base + 4), PF(base + 3), 1, 1, nullptr, nullptr, nullptr, nullptr);
}

static float* stem_fw(Stack& ws, hipStream_t st, void* const* din, int base,
                      const float* x6) {
  float* a1 = conv2d(ws, st, x6, 6, 192, 192, PF(base + 2), 64, 3, 3, 1, 1, PF(base + 1), PF(base + 0), 1, 1, nullptr, nullptr, nullptr, nullptr);
  float* a2 = conv2d(ws, st, a1, 64, 192, 192, PF(base + 5), 64, 3, 3, 1, 1, PF(base + 4), PF(base + 3), 1, 1, nullptr, nullptr, nullptr, nullptr);
  return conv2d(ws, st, a2, 64, 192, 192, PF(base + 6), 3, 1, 1, 1, 0, nullptr, nullptr, 0, 0, nullptr, nullptr, nullptr, nullptr);
}

extern "C" void kernel_launch(void* const* d_in, const int* in_sizes, int n_in,
                              void* d_out, int out_size, void* d_ws,
                              size_t ws_size, hipStream_t stream) {
  (void)in_sizes; (void)n_in; (void)out_size;
  hipStream_t st = stream;
  void* const* din = d_in;
  Stack ws{(char*)d_ws, 0, ws_size};

  // pytree (sorted-key) parameter index bases
  int c = 4;
  int ATTN[4]; for (int i = 0; i < 4; ++i) { ATTN[i] = c; c += 12; }
  int ENC = c; c += 108;
  int FFM21 = c; c += 9; int FFM32 = c; c += 9; int FFM43 = c; c += 9;
  int HEADB = c; c += 5;
  int MASKB = c; c += 8;
  int PROJ[4]; for (int i = 0; i < 4; ++i) { PROJ[i] = c; c += 3; }
  int STEMH = c; c += 7; int STEML = c; c += 7;

  const float* image  = PF(0);
  const float* sparse = PF(1);
  const float* est    = PF(2);
  const float* enorm  = PF(3);
  const int HW = 192 * 192;

  // Poisson pseudo-depth (persistent workgroup, LDS-resident Jacobi)
  float* eb = (float*)salloc(ws, (size_t)HW * 4);
  float* kn = (float*)salloc(ws, (size_t)HW * 4);
  float* vk = (float*)salloc(ws, (size_t)HW * 4);
  float* pseudo = (float*)salloc(ws, (size_t)HW * 4);
  k_poisson<<<dim3(1), dim3(1024), 2 * PNP * sizeof(float), st>>>(sparse, enorm, eb, kn, vk, pseudo);

  // stems
  float* xH = (float*)salloc(ws, 6LL * HW * 4);
  float* xL = (float*)salloc(ws, 6LL * HW * 4);
  k_pack6<<<GRID1D(6LL * HW), 256, 0, st>>>(xH, image, est, sparse, HW);
  k_pack6<<<GRID1D(6LL * HW), 256, 0, st>>>(xL, image, pseudo, sparse, HW);
  float* zL = stem_fw(ws, st, din, STEML, xL);
  float* zH = stem_fw(ws, st, din, STEMH, xH);

  // shared-weight encoders
  float* cL[4]; float* cH[4];
  run_encoder(ws, st, din, ENC, zL, cL);
  run_encoder(ws, st, din, ENC, zH, cH);

  // attention interaction + sigmoid gating per scale
  const int Cs[4] = {64, 128, 256, 512};
  const int Ssp[4] = {48, 24, 12, 6};
  float* fs[4];
  for (int i = 0; i < 4; ++i) {
    int Ci = Cs[i], s = Ssp[i], HWl = s * s;
    float* outL = (float*)salloc(ws, (size_t)Ci * HWl * 4);
    float* outH = (float*)salloc(ws, (size_t)Ci * HWl * 4);
    fs[i] = (float*)salloc(ws, (size_t)Ci * HWl * 4);
    size_t mark = ws.off;
    attn_block(ws, st, din, ATTN[i], cL[i], cH[i], Ci, s, outL, outH);
    ws.off = mark;
    k_gate<<<GRID1D(HWl), 256, 0, st>>>(fs[i], outL, outH, PF(MASKB + 2 * i + 1), PF(MASKB + 2 * i + 0), Ci, HWl);
  }

  // decoder
  float* p1 = conv2d(ws, st, fs[0], 64, 48, 48, PF(PROJ[0] + 2), 64, 1, 1, 1, 0, PF(PROJ[0] + 1), PF(PROJ[0] + 0), 1, 1, nullptr, nullptr, nullptr, nullptr);
  float* p2 = conv2d(ws, st, fs[1], 128, 24, 24, PF(PROJ[1] + 2), 128, 1, 1, 1, 0, PF(PROJ[1] + 1), PF(PROJ[1] + 0), 1, 1, nullptr, nullptr, nullptr, nullptr);
  float* p3 = conv2d(ws, st, fs[2], 256, 12, 12, PF(PROJ[2] + 2), 256, 1, 1, 1, 0, PF(PROJ[2] + 1), PF(PROJ[2] + 0), 1, 1, nullptr, nullptr, nullptr, nullptr);
  float* p4 = conv2d(ws, st, fs[3], 512, 6, 6, PF(PROJ[3] + 2), 256, 1, 1, 1, 0, PF(PROJ[3] + 1), PF(PROJ[3] + 0), 1, 1, nullptr, nullptr, nullptr, nullptr);

  float* u4 = (float*)salloc(ws, 256LL * 12 * 12 * 4);
  k_resize<<<GRID1D(256LL * 12 * 12), 256, 0, st>>>(u4, p4, 256, 6, 6, 12, 12);
  float* d43 = ffm_fw(ws, st, din, FFM43, p3, 256, u4, 256, 12, 256);
  float* u3 = (float*)salloc(ws, 256LL * 24 * 24 * 4);
  k_resize<<<GRID1D(256LL * 24 * 24), 256, 0, st>>>(u3, d43, 256, 12, 12, 24, 24);
  float* d32 = ffm_fw(ws, st, din, FFM32, p2, 128, u3, 256, 24, 128);
  float* u2 = (float*)salloc(ws, 128LL * 48 * 48 * 4);
  k_resize<<<GRID1D(128LL * 48 * 48), 256, 0, st>>>(u2, d32, 128, 24, 24, 48, 48);
  float* d21 = ffm_fw(ws, st, din, FFM21, p1, 64, u2, 128, 48, 64);

  float* hc = conv2d(ws, st, d21, 64, 48, 48, PF(HEADB + 3), 64, 3, 3, 1, 1, PF(HEADB + 2), PF(HEADB + 1), 1, 1, nullptr, nullptr, nullptr, nullptr);
  float* dd = conv2d(ws, st, hc, 64, 48, 48, PF(HEADB + 4), 1, 1, 1, 1, 0, nullptr, PF(HEADB + 0), 1, 0, nullptr, nullptr, nullptr, nullptr);
  k_resize<<<GRID1D((long long)HW), 256, 0, st>>>((float*)d_out, dd, 1, 48, 48, 192, 192);
}
